// ForgettingAttention_74775380623506
// MI455X (gfx1250) — matrix operand order repaired and checked by reading, not compiled, not run on hardware
//
#include <hip/hip_runtime.h>

typedef __attribute__((ext_vector_type(16))) _Float16 v16h;
typedef __attribute__((ext_vector_type(8)))  _Float16 v8h;
typedef __attribute__((ext_vector_type(4)))  _Float16 v4h;
typedef __attribute__((ext_vector_type(8)))  float    v8f;
typedef __attribute__((ext_vector_type(4)))  float    v4f;

#define B_  8
#define S_  2048
#define DK_ 128
#define DV_ 128

#define KSTRIDE 136   // halfs per K-tile row (128 + 8 pad) -> bank rotation
#define VSTRIDE 40    // halfs per V-tile row (32 + 8 pad)
#define PSTRIDE 40    // halfs per P row (32 + 8 pad)

// ---------------------------------------------------------------------------
// Pre-pass: per row, gate contributions qg = Q.Wq, kg = K.Wk, K -> f16
// row-major, V -> f16 transposed [b][dv][s]. One wave per row.
// ---------------------------------------------------------------------------
__global__ __launch_bounds__(256) void fa_prep(
    const float* __restrict__ Q, const float* __restrict__ K,
    const float* __restrict__ V, const float* __restrict__ Wq,
    const float* __restrict__ Wk,
    _Float16* __restrict__ K16, _Float16* __restrict__ VT16,
    float* __restrict__ qg, float* __restrict__ kg)
{
    const int lane = threadIdx.x & 31;
    const int w    = blockIdx.x * 8 + (threadIdx.x >> 5);
    const int b    = w >> 11;
    const int j    = w & (S_ - 1);

    const v4f q4  = *((const v4f*)(Q + (size_t)w * DK_) + lane);
    const v4f k4  = *((const v4f*)(K + (size_t)w * DK_) + lane);
    const v4f v4  = *((const v4f*)(V + (size_t)w * DK_) + lane);
    const v4f wq4 = ((const v4f*)Wq)[lane];
    const v4f wk4 = ((const v4f*)Wk)[lane];

    float qd = 0.f, kd = 0.f;
#pragma unroll
    for (int i = 0; i < 4; ++i) { qd += q4[i] * wq4[i]; kd += k4[i] * wk4[i]; }
#pragma unroll
    for (int m = 16; m >= 1; m >>= 1) {
        qd += __shfl_xor(qd, m, 32);
        kd += __shfl_xor(kd, m, 32);
    }
    if (lane == 0) { qg[w] = qd; kg[w] = kd; }

    v4h kh;
#pragma unroll
    for (int i = 0; i < 4; ++i) kh[i] = (_Float16)k4[i];
    *((v4h*)(K16 + (size_t)w * DK_) + lane) = kh;

    _Float16* vt = VT16 + (size_t)b * DV_ * S_ + j;
#pragma unroll
    for (int i = 0; i < 4; ++i)
        vt[(size_t)(lane * 4 + i) * S_] = (_Float16)v4[i];
}

// ---------------------------------------------------------------------------
// Helpers
// ---------------------------------------------------------------------------
__device__ __forceinline__ v16h ld_frag_lds(const _Float16* p)
{
    const v8h a = *(const v8h*)p;
    const v8h b = *(const v8h*)(p + 16);
    v16h r;
#pragma unroll
    for (int e = 0; e < 8; ++e) { r[e] = a[e]; r[8 + e] = b[e]; }
    return r;
}

// Issue async global->LDS copies for one 32-key tile of K (row major, padded)
// and V^T (dv-major, padded). 128 threads x 8 chunks x 16B = 16 KB.
__device__ __forceinline__ void issue_tile_async(
    const _Float16* __restrict__ Kb, const _Float16* __restrict__ Vb,
    _Float16* kt, _Float16* vt, int jbase, int t)
{
#pragma unroll
    for (int i = 0; i < 4; ++i) {
        const int ch = t + 128 * i;                 // 512 chunks of 16B
        const int r  = ch >> 4;                     // key row 0..31
        const int co = (ch & 15) * 8;               // half offset in row
        const _Float16* src = Kb + (size_t)(jbase + r) * DK_ + co;
        const unsigned dst = (unsigned)(size_t)(kt + r * KSTRIDE + co);
        asm volatile("global_load_async_to_lds_b128 %0, %1, off"
                     :: "v"(dst), "v"((unsigned long long)(size_t)src)
                     : "memory");
    }
#pragma unroll
    for (int i = 0; i < 4; ++i) {
        const int ch = t + 128 * i;
        const int dv = ch >> 2;                     // dv row 0..127
        const int co = (ch & 3) * 8;                // half offset in row
        const _Float16* src = Vb + (size_t)dv * S_ + jbase + co;
        const unsigned dst = (unsigned)(size_t)(vt + dv * VSTRIDE + co);
        asm volatile("global_load_async_to_lds_b128 %0, %1, off"
                     :: "v"(dst), "v"((unsigned long long)(size_t)src)
                     : "memory");
    }
}

// ---------------------------------------------------------------------------
// Main flash-attention kernel. 4 waves/block; each wave owns a 16-row query
// tile. K/V tiles staged once per block into LDS via async copies, double
// buffered; matmuls via v_wmma_f32_16x16x32_f16.
// ---------------------------------------------------------------------------
__global__ __launch_bounds__(128) void fa_main(
    const float* __restrict__ Q,
    const _Float16* __restrict__ K16, const _Float16* __restrict__ VT16,
    const float* __restrict__ qg, const float* __restrict__ kg,
    const float* __restrict__ bias, float* __restrict__ Out)
{
    __shared__ _Float16 ktile[2][32 * KSTRIDE];    // 2 x 8.5 KB
    __shared__ _Float16 vtile[2][128 * VSTRIDE];   // 2 x 10 KB
    __shared__ _Float16 plds[4][16 * PSTRIDE];     // 4 x 1.25 KB

    const int t     = threadIdx.x;
    const int lane  = t & 31;
    const int wid   = t >> 5;
    const int b     = blockIdx.y;
    const int qbase = (blockIdx.x * 4 + wid) * 16;
    const int n     = lane & 15;
    const int halfi = lane >> 4;
    const int base0 = halfi * 8;
    const float scale = 0.0883883476483184f;       // 1/sqrt(128)
    const float bv    = bias[0];

    const _Float16* Kb = K16  + (size_t)b * S_  * DK_;
    const _Float16* Vb = VT16 + (size_t)b * DV_ * S_;

    // ---- Q A-fragments ---------------------------------------------------
    v16h aq[4];
    {
        const float* qrow = Q + (size_t)(b * S_ + qbase + n) * DK_;
#pragma unroll
        for (int kk = 0; kk < 4; ++kk) {
            const float* qp = qrow + kk * 32 + base0;
#pragma unroll
            for (int e = 0; e < 8; ++e) aq[kk][e]     = (_Float16)qp[e];
#pragma unroll
            for (int e = 0; e < 8; ++e) aq[kk][8 + e] = (_Float16)qp[16 + e];
        }
    }

    float qgb[8];
#pragma unroll
    for (int c = 0; c < 8; ++c) qgb[c] = qg[b * S_ + qbase + c + base0] + bv;

    v8f O[8] = {};
    float mrow[8], lrow[8];
#pragma unroll
    for (int c = 0; c < 8; ++c) { mrow[c] = -1e30f; lrow[c] = 0.f; }

    _Float16* pl = &plds[wid][0];

    const int nt   = blockIdx.x * 2 + 2;           // j-tiles this block needs
    const int myNt = (qbase + 16 + 31) >> 5;       // j-tiles this wave needs

    issue_tile_async(Kb, Vb, ktile[0], vtile[0], 0, t);

    for (int jt = 0; jt < nt; ++jt) {
        const int jbase = jt * 32;

        if (jt + 1 < nt) {
            issue_tile_async(Kb, Vb, ktile[(jt + 1) & 1], vtile[(jt + 1) & 1],
                             jbase + 32, t);
            asm volatile("s_wait_asynccnt 0x8" ::: "memory");
        } else {
            asm volatile("s_wait_asynccnt 0x0" ::: "memory");
        }
        __syncthreads();                           // tile jt ready for all waves

        if (jt < myNt) {                           // wave-uniform causal skip
            const _Float16* kt = ktile[jt & 1];
            const _Float16* vt = vtile[jt & 1];

            // ---- S = Q K^T (8 WMMAs) ------------------------------------
            v8f s0 = {}, s1 = {};
            const _Float16* k0p = kt + (size_t)n * KSTRIDE + halfi * 8;
            const _Float16* k1p = k0p + 16 * KSTRIDE;
#pragma unroll
            for (int kk = 0; kk < 4; ++kk) {
                const v16h kb = ld_frag_lds(k0p + kk * 32);
                s0 = __builtin_amdgcn_wmma_f32_16x16x32_f16(false, aq[kk], false, kb,
                                                            (short)0, s0, false, false);
            }
#pragma unroll
            for (int kk = 0; kk < 4; ++kk) {
                const v16h kb = ld_frag_lds(k1p + kk * 32);
                s1 = __builtin_amdgcn_wmma_f32_16x16x32_f16(false, aq[kk], false, kb,
                                                            (short)0, s1, false, false);
            }

            // ---- gate + causal mask + online softmax --------------------
            const float kg0 = kg[b * S_ + jbase + n];
            const float kg1 = kg[b * S_ + jbase + 16 + n];
#pragma unroll
            for (int c = 0; c < 8; ++c) {
                const int rowr = qbase + c + base0;
                const float g0 = 1.f / (1.f + __expf(-(qgb[c] + kg0)));
                const float g1 = 1.f / (1.f + __expf(-(qgb[c] + kg1)));
                float x0 = s0[c] * scale * g0;
                float x1 = s1[c] * scale * g1;
                x0 = (jbase + n      > rowr) ? -1e30f : x0;
                x1 = (jbase + 16 + n > rowr) ? -1e30f : x1;

                float tm = fmaxf(x0, x1);
                tm = fmaxf(tm, __shfl_xor(tm, 1, 32));
                tm = fmaxf(tm, __shfl_xor(tm, 2, 32));
                tm = fmaxf(tm, __shfl_xor(tm, 4, 32));
                tm = fmaxf(tm, __shfl_xor(tm, 8, 32));

                const float nm = fmaxf(mrow[c], tm);
                const float al = __expf(mrow[c] - nm);
                mrow[c] = nm;
                const float p0 = __expf(x0 - nm);
                const float p1 = __expf(x1 - nm);
                float rs = p0 + p1;
                rs += __shfl_xor(rs, 1, 32);
                rs += __shfl_xor(rs, 2, 32);
                rs += __shfl_xor(rs, 4, 32);
                rs += __shfl_xor(rs, 8, 32);
                lrow[c] = lrow[c] * al + rs;

#pragma unroll
                for (int nn = 0; nn < 8; ++nn) O[nn][c] *= al;

                pl[(c + base0) * PSTRIDE + n]      = (_Float16)p0;
                pl[(c + base0) * PSTRIDE + 16 + n] = (_Float16)p1;
            }

            // ---- P as A-fragment (wave-private LDS round trip) ----------
            v16h pa;
            {
                const v8h pal = *(const v8h*)(pl + n * PSTRIDE + base0);
                const v8h pah = *(const v8h*)(pl + n * PSTRIDE + base0 + 16);
#pragma unroll
                for (int e = 0; e < 8; ++e) { pa[e] = pal[e]; pa[8 + e] = pah[e]; }
            }

            // ---- O += P V (8 WMMAs) -------------------------------------
#pragma unroll
            for (int nn = 0; nn < 8; ++nn) {
                const v16h vb = ld_frag_lds(vt + (size_t)(nn * 16 + n) * VSTRIDE
                                               + halfi * 8);
                O[nn] = __builtin_amdgcn_wmma_f32_16x16x32_f16(false, pa, false, vb,
                                                               (short)0, O[nn],
                                                               false, false);
            }
        }

        __syncthreads();   // all waves done reading buffer jt before overwrite
    }

    // ---- normalize and store --------------------------------------------
#pragma unroll
    for (int c = 0; c < 8; ++c) {
        const float inv = 1.f / lrow[c];
        float* orow = Out + (size_t)(b * S_ + qbase + c + base0) * DV_ + n;
#pragma unroll
        for (int nn = 0; nn < 8; ++nn) orow[nn * 16] = O[nn][c] * inv;
    }
}

// ---------------------------------------------------------------------------
extern "C" void kernel_launch(void* const* d_in, const int* in_sizes, int n_in,
                              void* d_out, int out_size, void* d_ws, size_t ws_size,
                              hipStream_t stream)
{
    (void)in_sizes; (void)n_in; (void)out_size; (void)ws_size;
    const float* Q    = (const float*)d_in[0];
    const float* K    = (const float*)d_in[1];
    const float* V    = (const float*)d_in[2];
    const float* Wq   = (const float*)d_in[3];
    const float* Wk   = (const float*)d_in[4];
    const float* bias = (const float*)d_in[5];
    // d_in[6] (mask) unused: causal mask computed analytically.

    char* ws = (char*)d_ws;
    const size_t kvBytes = (size_t)B_ * S_ * DK_ * sizeof(_Float16);   // 4 MiB
    _Float16* K16  = (_Float16*)ws;
    _Float16* VT16 = (_Float16*)(ws + kvBytes);
    float*    qg   = (float*)(ws + 2 * kvBytes);
    float*    kg   = qg + (size_t)B_ * S_;
    float*    out  = (float*)d_out;

    fa_prep<<<dim3(B_ * S_ / 8), dim3(256), 0, stream>>>(Q, K, V, Wq, Wk,
                                                         K16, VT16, qg, kg);
    fa_main<<<dim3(S_ / 64, B_), dim3(128), 0, stream>>>(Q, K16, VT16, qg, kg,
                                                         bias, out);
}
